// CRF_10058813407510
// MI455X (gfx1250) — matrix operand order repaired and checked by reading, not compiled, not run on hardware
//
#include <hip/hip_runtime.h>
#include <hip/hip_bf16.h>

typedef __attribute__((ext_vector_type(16))) _Float16 v16h;
typedef __attribute__((ext_vector_type(8)))  _Float16 v8h;
typedef __attribute__((ext_vector_type(8)))  float    v8f;

namespace {
constexpr int Hh = 96, Ww = 96, Nn = Hh * Ww;   // 9216 pixels
constexpr int Ll = 21, LP = 32;                 // labels, padded labels
constexpr int R  = 9,  T  = 2 * R + 1;          // gaussian radius/taps (sigma=3)
constexpr float POS_W = 3.0f, BI_W = 10.0f;
}

// exp(-0.5*(d/3)^2), d = -9..9  (unnormalized separable spatial kernel)
__constant__ float c_kern[T] = {
  0.0111090f, 0.0285656f, 0.0657282f, 0.1353353f, 0.2493522f,
  0.4111123f, 0.6065307f, 0.8007374f, 0.9459594f, 1.0000000f,
  0.9459594f, 0.8007374f, 0.6065307f, 0.4111123f, 0.2493522f,
  0.1353353f, 0.0657282f, 0.0285656f, 0.0111090f };

// ---------------------------------------------------------------- features + nsp
__global__ void k_init(const float* __restrict__ I, float* __restrict__ feats,
                       float* __restrict__ nsp) {
  int n = blockIdx.x * blockDim.x + threadIdx.x;
  if (n >= Nn) return;
  int y = n / Ww, x = n - y * Ww;
  feats[n * 8 + 0] = (float)x * (1.0f / 80.0f);          // x / BI_X_STD
  feats[n * 8 + 1] = (float)y * (1.0f / 80.0f);          // y / BI_Y_STD
  feats[n * 8 + 2] = I[0 * Nn + n] * (1.0f / 13.0f);
  feats[n * 8 + 3] = I[1 * Nn + n] * (1.0f / 13.0f);
  feats[n * 8 + 4] = I[2 * Nn + n] * (1.0f / 13.0f);
  feats[n * 8 + 5] = 0.0f; feats[n * 8 + 6] = 0.0f; feats[n * 8 + 7] = 0.0f;
  // nsp = rsqrt(conv2d(ones)) ; conv of ones with separable kernel = sy(y)*sx(x)
  float sy = 0.0f, sx = 0.0f;
  #pragma unroll
  for (int d = -R; d <= R; ++d) {
    float w = c_kern[d + R];
    if ((unsigned)(y + d) < (unsigned)Hh) sy += w;
    if ((unsigned)(x + d) < (unsigned)Ww) sx += w;
  }
  nsp[n] = rsqrtf(sy * sx);
}

// ------------------------------------------------- K (fp16, 162MB) + nbi rowsums
__global__ void k_buildK(const float* __restrict__ feats, _Float16* __restrict__ Kh,
                         float* __restrict__ nbi) {
  const int n = blockIdx.x;
  const int tid = threadIdx.x;
  const float f0 = feats[n * 8 + 0], f1 = feats[n * 8 + 1], f2 = feats[n * 8 + 2];
  const float f3 = feats[n * 8 + 3], f4 = feats[n * 8 + 4];
  float sum = 0.0f;
  for (int m = tid; m < Nn; m += 256) {
    float d0 = f0 - feats[m * 8 + 0];
    float d1 = f1 - feats[m * 8 + 1];
    float d2 = f2 - feats[m * 8 + 2];
    float d3 = f3 - feats[m * 8 + 3];
    float d4 = f4 - feats[m * 8 + 4];
    float D = d0 * d0 + d1 * d1 + d2 * d2 + d3 * d3 + d4 * d4;
    float kv = __expf(-0.5f * D);
    Kh[(size_t)n * Nn + m] = (_Float16)kv;
    sum += kv;
  }
  __shared__ float s[256];
  s[tid] = sum;
  __syncthreads();
  for (int st = 128; st > 0; st >>= 1) {
    if (tid < st) s[tid] += s[tid + st];
    __syncthreads();
  }
  if (tid == 0) nbi[n] = rsqrtf(s[0]);
}

// ------------------------------------------ V[l][m] = nbi*Q (fp16, label-major)
__global__ void k_buildV(const float* __restrict__ Q, const float* __restrict__ nbi,
                         _Float16* __restrict__ Vh) {
  int idx = blockIdx.x * blockDim.x + threadIdx.x;
  if (idx >= LP * Nn) return;
  int l = idx / Nn, m = idx - l * Nn;
  float v = (l < Ll) ? nbi[m] * Q[l * Nn + m] : 0.0f;
  Vh[idx] = (_Float16)v;
}

// ------------------------------------------------------------ separable spatial
__global__ void k_conv_y(const float* __restrict__ Q, const float* __restrict__ nsp,
                         float* __restrict__ tmp) {
  int idx = blockIdx.x * blockDim.x + threadIdx.x;
  if (idx >= Ll * Nn) return;
  int l = idx / Nn, n = idx - l * Nn;
  int y = n / Ww, x = n - y * Ww;
  float s = 0.0f;
  #pragma unroll
  for (int d = -R; d <= R; ++d) {
    int yy = y + d;
    if ((unsigned)yy < (unsigned)Hh) {
      int nn = yy * Ww + x;
      s += c_kern[d + R] * nsp[nn] * Q[l * Nn + nn];
    }
  }
  tmp[idx] = s;
}

__global__ void k_conv_x(const float* __restrict__ tmp, const float* __restrict__ nsp,
                         float* __restrict__ spm) {
  int idx = blockIdx.x * blockDim.x + threadIdx.x;
  if (idx >= Ll * Nn) return;
  int l = idx / Nn, n = idx - l * Nn;
  int y = n / Ww, x = n - y * Ww;
  float s = 0.0f;
  #pragma unroll
  for (int d = -R; d <= R; ++d) {
    int xx = x + d;
    if ((unsigned)xx < (unsigned)Ww) s += c_kern[d + R] * tmp[l * Nn + y * Ww + xx];
  }
  spm[idx] = POS_W * nsp[n] * s;
}

// ------------------------------------------------- bilateral: bi = BI_W*nbi*(K@V)
// 576 row-tiles of 16 pixels; 4 waves/block split the m-loop; WMMA f16->f32.
__global__ void __launch_bounds__(128) k_bilateral(
    const _Float16* __restrict__ Kh, const _Float16* __restrict__ Vh,
    const float* __restrict__ nbi, float* __restrict__ bim) {
  const int n0   = blockIdx.x * 16;
  const int tid  = threadIdx.x;
  const int wave = tid >> 5;
  const int lane = tid & 31;
  const int hsel = lane >> 4;        // 0: lanes 0-15, 1: lanes 16-31
  const int lr   = lane & 15;
  const int kbA  = hsel * 16;        // A layout: 16 contiguous K per lane
  const int kbB  = hsel * 16;        // B layout: 16 contiguous K per lane

  const _Float16* Krow = Kh + (size_t)(n0 + lr) * Nn;  // A row = pixel n0+lr
  const _Float16* V0   = Vh + (size_t)lr * Nn;         // B col = label lr
  const _Float16* V1   = Vh + (size_t)(16 + lr) * Nn;  // B col = label 16+lr

  v8f acc0 = {}; v8f acc1 = {};
  for (int m0 = wave * 32; m0 < Nn; m0 += 128) {
    v8h alo = *(const v8h*)(Krow + m0 + kbA);
    v8h ahi = *(const v8h*)(Krow + m0 + kbA + 8);
    v16h a = __builtin_shufflevector(alo, ahi, 0, 1, 2, 3, 4, 5, 6, 7,
                                     8, 9, 10, 11, 12, 13, 14, 15);
    v16h b0 = *(const v16h*)(V0 + m0 + kbB);
    v16h b1 = *(const v16h*)(V1 + m0 + kbB);
    acc0 = __builtin_amdgcn_wmma_f32_16x16x32_f16(false, a, false, b0,
                                                  (short)0, acc0, false, false);
    acc1 = __builtin_amdgcn_wmma_f32_16x16x32_f16(false, a, false, b1,
                                                  (short)0, acc1, false, false);
  }

  __shared__ float red[4][32][16];
  #pragma unroll
  for (int j = 0; j < 8; ++j) {
    red[wave][lane][j]     = acc0[j];
    red[wave][lane][j + 8] = acc1[j];
  }
  __syncthreads();
  if (wave == 0) {
    #pragma unroll
    for (int j = 0; j < 16; ++j) {
      float s = red[0][lane][j] + red[1][lane][j] + red[2][lane][j] + red[3][lane][j];
      int v = j & 7, lt = j >> 3;
      int M = v + hsel * 8;            // C/D layout: lane>=16 holds rows 8..15
      int l = lt * 16 + lr;            // column = label
      if (l < Ll) bim[(size_t)l * Nn + n0 + M] = BI_W * nbi[n0 + M] * s;
    }
  }
}

// ---------------------------------------------------- softmax(-U [+ sp + bi])
__global__ void k_update(const float* __restrict__ U, const float* __restrict__ spm,
                         const float* __restrict__ bim, float* __restrict__ Qout,
                         int use_msgs) {
  int n = blockIdx.x * blockDim.x + threadIdx.x;
  if (n >= Nn) return;
  float lg[Ll];
  float mx = -3.4e38f;
  #pragma unroll
  for (int l = 0; l < Ll; ++l) {
    float v = -U[l * Nn + n];
    if (use_msgs) v += spm[l * Nn + n] + bim[l * Nn + n];
    lg[l] = v;
    mx = fmaxf(mx, v);
  }
  float s = 0.0f;
  #pragma unroll
  for (int l = 0; l < Ll; ++l) { float e = __expf(lg[l] - mx); lg[l] = e; s += e; }
  float inv = 1.0f / s;
  #pragma unroll
  for (int l = 0; l < Ll; ++l) Qout[l * Nn + n] = lg[l] * inv;
}

extern "C" void kernel_launch(void* const* d_in, const int* in_sizes, int n_in,
                              void* d_out, int out_size, void* d_ws, size_t ws_size,
                              hipStream_t stream) {
  (void)in_sizes; (void)n_in; (void)out_size; (void)ws_size;
  const float* U = (const float*)d_in[0];   // [1,21,96,96]
  const float* I = (const float*)d_in[1];   // [1,3,96,96]
  float* out = (float*)d_out;               // [1,21,96,96]

  char* ws = (char*)d_ws;
  size_t off = 0;
  auto take = [&](size_t bytes) -> char* {
    char* p = ws + off;
    off += (bytes + 255) & ~(size_t)255;
    return p;
  };
  _Float16* Kh  = (_Float16*)take((size_t)Nn * Nn * sizeof(_Float16)); // 162 MB
  _Float16* Vh  = (_Float16*)take((size_t)LP * Nn * sizeof(_Float16));
  float* feats  = (float*)take((size_t)Nn * 8 * sizeof(float));
  float* nbi    = (float*)take((size_t)Nn * sizeof(float));
  float* nsp    = (float*)take((size_t)Nn * sizeof(float));
  float* Qbuf   = (float*)take((size_t)Ll * Nn * sizeof(float));
  float* tmp    = (float*)take((size_t)Ll * Nn * sizeof(float));
  float* spm    = (float*)take((size_t)Ll * Nn * sizeof(float));
  float* bim    = (float*)take((size_t)Ll * Nn * sizeof(float));

  dim3 b256(256);
  k_init<<<(Nn + 255) / 256, b256, 0, stream>>>(I, feats, nsp);
  k_buildK<<<Nn, b256, 0, stream>>>(feats, Kh, nbi);
  k_update<<<(Nn + 255) / 256, b256, 0, stream>>>(U, spm, bim, Qbuf, 0);

  for (int it = 0; it < 10; ++it) {
    k_buildV<<<(LP * Nn + 255) / 256, b256, 0, stream>>>(Qbuf, nbi, Vh);
    k_conv_y<<<(Ll * Nn + 255) / 256, b256, 0, stream>>>(Qbuf, nsp, tmp);
    k_conv_x<<<(Ll * Nn + 255) / 256, b256, 0, stream>>>(tmp, nsp, spm);
    k_bilateral<<<Nn / 16, dim3(128), 0, stream>>>(Kh, Vh, nbi, bim);
    float* Qdst = (it == 9) ? out : Qbuf;
    k_update<<<(Nn + 255) / 256, b256, 0, stream>>>(U, spm, bim, Qdst, 1);
  }
}
